// miniGPT_34368328303209
// MI455X (gfx1250) — compile-verified, matrix operand order checked by reading, not run on hardware
//
#include <hip/hip_runtime.h>
#include <math.h>

// Problem dims (compile-time constants from the reference)
#define LL   4
#define DM   768
#define HH   12
#define HDD  64
#define HIDD 3072
#define VV   32000
#define BB   4
#define SS   1024
#define MM   (BB * SS)   // 4096 token rows
#define KC   64          // GEMM K-chunk (two 16x16x32 WMMA K-steps)

typedef __bf16 bf16;
typedef __attribute__((ext_vector_type(16))) __bf16 bf16x16;
typedef __attribute__((ext_vector_type(8)))  __bf16 bf16x8;
typedef __attribute__((ext_vector_type(8)))  float  f32x8;

// K index for element e of an A/B fragment (16-bit data, 16x16x32):
//   K = (e&7) + (e>>3)*16 + 8*khalf
__device__ __forceinline__ int kmap(int e, int khalf) {
    return (e & 7) + ((e >> 3) << 4) + (khalf << 3);
}

// Fragment from a row-major row pointer (global): two contiguous 16B chunks.
__device__ __forceinline__ bf16x16 load_frag(const bf16* rowp, int k0, int khalf) {
    const bf16* p = rowp + k0 + (khalf << 3);
    bf16x8 lo = *reinterpret_cast<const bf16x8*>(p);
    bf16x8 hi = *reinterpret_cast<const bf16x8*>(p + 16);
    bf16x16 f;
#pragma unroll
    for (int i = 0; i < 8; ++i) { f[i] = lo[i]; f[i + 8] = hi[i]; }
    return f;
}

// Fragment from an LDS row pointer -> two ds_load_b128.
__device__ __forceinline__ bf16x16 lds_frag(const bf16* rowp, int khalf) {
    const bf16* p = rowp + (khalf << 3);
    bf16x8 lo = *reinterpret_cast<const bf16x8*>(p);
    bf16x8 hi = *reinterpret_cast<const bf16x8*>(p + 16);
    bf16x16 f;
#pragma unroll
    for (int i = 0; i < 8; ++i) { f[i] = lo[i]; f[i + 8] = hi[i]; }
    return f;
}

__device__ __forceinline__ f32x8 wmma_bf16(bf16x16 a, bf16x16 b, f32x8 c) {
    return __builtin_amdgcn_wmma_f32_16x16x32_bf16(false, a, false, b,
                                                   (short)0, c, false, false);
}

// Async DMA: 16 bytes global -> LDS, tracked by ASYNCcnt (CDNA5 path).
// VDST operand carries the LDS byte offset (low 32 bits of the flat address).
__device__ __forceinline__ void async_cp16(bf16* dst_lds, const bf16* src_g) {
    unsigned lds_off = (unsigned)(size_t)dst_lds;
    asm volatile("global_load_async_to_lds_b128 %0, %1, off"
                 :: "v"(lds_off), "v"(src_g) : "memory");
}

__device__ __forceinline__ void wait_async0() {
    asm volatile("s_wait_asynccnt 0x0" ::: "memory");
}

__device__ __forceinline__ void store_f32(float v, float* p, bool nt) {
#if __has_builtin(__builtin_nontemporal_store)
    if (nt) { __builtin_nontemporal_store(v, p); return; }
#endif
    *p = v;
}

// ---------------------------------------------------------------- embedding
__global__ void embed_kernel(const int* __restrict__ tokens,
                             const float* __restrict__ emb,
                             const float* __restrict__ pe,
                             float* __restrict__ x) {
    const int tid = blockIdx.x * blockDim.x + threadIdx.x;
    if (tid >= MM * DM) return;
    const int d  = tid % DM;
    const int bs = tid / DM;
    const int s  = bs % SS;
    x[tid] = emb[(size_t)tokens[bs] * DM + d] + pe[(size_t)s * DM + d];
}

// ------------------------------------------- f32 -> bf16 transpose ([K,N] -> [N,K])
__global__ void convT_kernel(const float* __restrict__ w, bf16* __restrict__ wT,
                             int K, int N) {
    const int tid = blockIdx.x * blockDim.x + threadIdx.x;
    if (tid >= K * N) return;
    const int kk = tid / N;
    const int nn = tid % N;
    wT[(size_t)nn * K + kk] = (bf16)w[tid];
}

// ---------------------------------------------------------------- layernorm
__global__ void ln_kernel(const float* __restrict__ x,
                          const float* __restrict__ g,
                          const float* __restrict__ b,
                          bf16* __restrict__ out) {
    __shared__ float red[256];
    const int row = blockIdx.x;
    const float* xr = x + (size_t)row * DM;

    float s = 0.f;
    for (int i = threadIdx.x; i < DM; i += 256) s += xr[i];
    red[threadIdx.x] = s;
    __syncthreads();
    for (int st = 128; st > 0; st >>= 1) {
        if (threadIdx.x < st) red[threadIdx.x] += red[threadIdx.x + st];
        __syncthreads();
    }
    const float mu = red[0] * (1.0f / DM);
    __syncthreads();

    float v = 0.f;
    for (int i = threadIdx.x; i < DM; i += 256) {
        const float d = xr[i] - mu;
        v += d * d;
    }
    red[threadIdx.x] = v;
    __syncthreads();
    for (int st = 128; st > 0; st >>= 1) {
        if (threadIdx.x < st) red[threadIdx.x] += red[threadIdx.x + st];
        __syncthreads();
    }
    const float rstd = rsqrtf(red[0] * (1.0f / DM) + 1e-5f);

    bf16* orow = out + (size_t)row * DM;
    for (int i = threadIdx.x; i < DM; i += 256)
        orow[i] = (bf16)((xr[i] - mu) * rstd * g[i] + b[i]);
}

// -------------------------------------------------------- generic WMMA GEMM
// C[M,N] = A[M,K](bf16) * BT[N,K](bf16)^T, fusion flags are template params.
// Block: 256 threads = 8 waves (2 in M x 4 in N); block tile 64x128.
// Each wave computes a 32x32 tile: 8 WMMAs per 64-wide K-chunk from LDS.
// A/B K-chunks staged to LDS with async DMA, double buffered (ASYNCcnt).
template<bool BIAS, bool RES, bool RELU, bool WF, bool WB, bool NTST>
__global__ void __launch_bounds__(256) gemm_bf16_kernel(
    const bf16* __restrict__ A, const bf16* __restrict__ BT,
    const float* __restrict__ bias, const float* __restrict__ resid,
    float* __restrict__ outF, bf16* __restrict__ outB,
    int M, int N, int K) {
    __shared__ bf16 Ab[2][64][KC];    // 16 KB
    __shared__ bf16 Bb[2][128][KC];   // 32 KB

    const int tid   = threadIdx.x;
    const int lane  = tid & 31;
    const int wave  = tid >> 5;
    const int l16   = lane & 15;
    const int khalf = lane >> 4;
    const int mw    = (wave & 1) * 32;   // wave row base within block tile
    const int nw    = (wave >> 1) * 32;  // wave col base within block tile
    const int mBase = blockIdx.x * 64;
    const int nBase = blockIdx.y * 128;

    // Staging: 16B segments; A tile = 512 segs (2/thread), B tile = 1024 (4/thread)
    auto stage = [&](int buf, int k0) {
#pragma unroll
        for (int i = 0; i < 2; ++i) {
            const int s = tid + i * 256;
            const int r = s >> 3, c = (s & 7) * 8;
            async_cp16(&Ab[buf][r][c], A + (size_t)(mBase + r) * K + k0 + c);
        }
#pragma unroll
        for (int i = 0; i < 4; ++i) {
            const int s = tid + i * 256;
            const int r = s >> 3, c = (s & 7) * 8;
            async_cp16(&Bb[buf][r][c], BT + (size_t)(nBase + r) * K + k0 + c);
        }
    };

    f32x8 acc[2][2] = {};
    stage(0, 0);

    int cur = 0;
    for (int k0 = 0; k0 < K; k0 += KC) {
        wait_async0();      // current buffer's DMA (issued last iter) complete
        __syncthreads();    // visible to all waves; prev reads already consumed

        if (k0 + KC < K) stage(cur ^ 1, k0 + KC);

#pragma unroll
        for (int kk = 0; kk < KC; kk += 32) {
            bf16x16 a0 = lds_frag(&Ab[cur][mw + l16][kk],      khalf);
            bf16x16 a1 = lds_frag(&Ab[cur][mw + 16 + l16][kk], khalf);
            bf16x16 b0 = lds_frag(&Bb[cur][nw + l16][kk],      khalf);
            bf16x16 b1 = lds_frag(&Bb[cur][nw + 16 + l16][kk], khalf);
            acc[0][0] = wmma_bf16(a0, b0, acc[0][0]);
            acc[0][1] = wmma_bf16(a0, b1, acc[0][1]);
            acc[1][0] = wmma_bf16(a1, b0, acc[1][0]);
            acc[1][1] = wmma_bf16(a1, b1, acc[1][1]);
        }
        cur ^= 1;
    }

#pragma unroll
    for (int ai = 0; ai < 2; ++ai)
#pragma unroll
        for (int bi = 0; bi < 2; ++bi) {
            const int col = nBase + nw + bi * 16 + l16;
            const float bb = BIAS ? bias[col] : 0.f;
#pragma unroll
            for (int r = 0; r < 8; ++r) {
                const int row = mBase + mw + ai * 16 + r + (khalf << 3);
                const size_t idx = (size_t)row * N + col;
                float vv = acc[ai][bi][r];
                if (BIAS) vv += bb;
                if (RES)  vv += resid[idx];
                if (RELU) vv = fmaxf(vv, 0.f);
                if (WF)   store_f32(vv, &outF[idx], NTST);
                if (WB)   outB[idx] = (bf16)vv;
            }
        }
}

// ----------------------------------------------- causal flash-style attention
// One wave per (b, h, 16-query tile). q/k/v/out are bf16 [B,S,D] row-major.
__global__ void __launch_bounds__(32) attn_kernel(
    const bf16* __restrict__ q, const bf16* __restrict__ k,
    const bf16* __restrict__ v, bf16* __restrict__ out, float scale) {
    __shared__ float pT[16][32];

    const int lane  = threadIdx.x & 31;
    const int l16   = lane & 15;
    const int khalf = lane >> 4;
    const int qBase = blockIdx.x * 16;
    const int h     = blockIdx.y;
    const int b     = blockIdx.z;
    const size_t seqBase = (size_t)b * SS;

    const bf16* qRow = q + (seqBase + qBase + l16) * DM + h * HDD;
    const bf16x16 qf0 = load_frag(qRow, 0,  khalf);
    const bf16x16 qf1 = load_frag(qRow, 32, khalf);

    float mrow[8], lrow[8];
    f32x8 o[4] = {};
#pragma unroll
    for (int r = 0; r < 8; ++r) { mrow[r] = -3.0e38f; lrow[r] = 0.f; }

    for (int kc = 0; kc < qBase + 16; kc += 32) {
        // scores for 32 keys: two 16x16 C-tiles, K=64 over head dim (2 wmma each)
        const bf16* kRow0 = k + (seqBase + kc + l16) * DM + h * HDD;
        const bf16* kRow1 = kRow0 + (size_t)16 * DM;
        f32x8 s0 = {}, s1 = {};
        s0 = wmma_bf16(qf0, load_frag(kRow0, 0,  khalf), s0);
        s0 = wmma_bf16(qf1, load_frag(kRow0, 32, khalf), s0);
        s1 = wmma_bf16(qf0, load_frag(kRow1, 0,  khalf), s1);
        s1 = wmma_bf16(qf1, load_frag(kRow1, 32, khalf), s1);

        // online softmax update (row = r + 8*khalf within this q tile)
#pragma unroll
        for (int r = 0; r < 8; ++r) {
            const int m  = qBase + r + (khalf << 3);
            const int n0 = kc + l16;
            const int n1 = n0 + 16;
            const float a0 = (n0 <= m) ? s0[r] * scale : -3.0e38f;
            const float a1 = (n1 <= m) ? s1[r] * scale : -3.0e38f;

            float mx = fmaxf(a0, a1);
#pragma unroll
            for (int off = 1; off < 16; off <<= 1)
                mx = fmaxf(mx, __shfl_xor(mx, off, 16));
            const float mnew = fmaxf(mrow[r], mx);
            const float corr = __expf(mrow[r] - mnew);
            const float p0 = __expf(a0 - mnew);
            const float p1 = __expf(a1 - mnew);
            float sum = p0 + p1;
#pragma unroll
            for (int off = 1; off < 16; off <<= 1)
                sum += __shfl_xor(sum, off, 16);
            lrow[r] = lrow[r] * corr + sum;
            mrow[r] = mnew;
#pragma unroll
            for (int t = 0; t < 4; ++t) o[t][r] *= corr;

            pT[r + (khalf << 3)][l16]      = p0;
            pT[r + (khalf << 3)][16 + l16] = p1;
        }
        __syncthreads();

        // re-layout P (C-layout -> A-fragment) via LDS, then P @ V
        bf16x16 pf;
#pragma unroll
        for (int e = 0; e < 16; ++e)
            pf[e] = (bf16)pT[l16][kmap(e, khalf)];

#pragma unroll
        for (int t = 0; t < 4; ++t) {
            bf16x16 vf;
#pragma unroll
            for (int e = 0; e < 16; ++e) {
                const int key = kc + kmap(e, khalf);
                vf[e] = v[(seqBase + key) * DM + h * HDD + t * 16 + l16];
            }
            o[t] = wmma_bf16(pf, vf, o[t]);
        }
        __syncthreads();
    }

#pragma unroll
    for (int t = 0; t < 4; ++t)
#pragma unroll
        for (int r = 0; r < 8; ++r) {
            const int row = qBase + r + (khalf << 3);
            out[(seqBase + row) * DM + h * HDD + t * 16 + l16] =
                (bf16)(o[t][r] / lrow[r]);
        }
}

// ------------------------------------------------------------------- driver
extern "C" void kernel_launch(void* const* d_in, const int* in_sizes, int n_in,
                              void* d_out, int out_size, void* d_ws, size_t ws_size,
                              hipStream_t stream) {
    (void)in_sizes; (void)n_in; (void)out_size; (void)ws_size;

    const int*   tokens = (const int*)  d_in[0];
    const float* emb    = (const float*)d_in[1];
    const float* pe     = (const float*)d_in[2];
    const float* ln1_g  = (const float*)d_in[3];
    const float* ln1_b  = (const float*)d_in[4];
    const float* wq     = (const float*)d_in[5];
    const float* bq     = (const float*)d_in[6];
    const float* wk     = (const float*)d_in[7];
    const float* bk     = (const float*)d_in[8];
    const float* wv     = (const float*)d_in[9];
    const float* bv     = (const float*)d_in[10];
    const float* wo     = (const float*)d_in[11];
    const float* bo     = (const float*)d_in[12];
    const float* ln2_g  = (const float*)d_in[13];
    const float* ln2_b  = (const float*)d_in[14];
    const float* w1     = (const float*)d_in[15];
    const float* b1     = (const float*)d_in[16];
    const float* w2     = (const float*)d_in[17];
    const float* b2     = (const float*)d_in[18];
    const float* lnf_g  = (const float*)d_in[19];
    const float* lnf_b  = (const float*)d_in[20];
    const float* w_out  = (const float*)d_in[21];
    float* logits = (float*)d_out;

    char* ws = (char*)d_ws;
    size_t off = 0;
    auto take = [&](size_t bytes) -> void* {
        void* p = ws + off;
        off += (bytes + 255) & ~(size_t)255;
        return p;
    };
    float* xF  = (float*)take((size_t)MM * DM * 4);
    bf16*  hB  = (bf16*) take((size_t)MM * DM * 2);
    bf16*  qB  = (bf16*) take((size_t)MM * DM * 2);
    bf16*  kB  = (bf16*) take((size_t)MM * DM * 2);
    bf16*  vB  = (bf16*) take((size_t)MM * DM * 2);
    bf16*  aB  = (bf16*) take((size_t)MM * DM * 2);
    bf16*  fB  = (bf16*) take((size_t)MM * HIDD * 2);
    bf16*  wqT = (bf16*) take((size_t)LL * DM * DM * 2);
    bf16*  wkT = (bf16*) take((size_t)LL * DM * DM * 2);
    bf16*  wvT = (bf16*) take((size_t)LL * DM * DM * 2);
    bf16*  woT = (bf16*) take((size_t)LL * DM * DM * 2);
    bf16*  w1T = (bf16*) take((size_t)LL * DM * HIDD * 2);
    bf16*  w2T = (bf16*) take((size_t)LL * HIDD * DM * 2);
    bf16*  wOT = (bf16*) take((size_t)DM * VV * 2);

    auto cdiv = [](int a, int b) { return (a + b - 1) / b; };

    // Weight conversion: f32 [K,N] -> bf16 [N,K] (transposed for contiguous B frags)
    for (int l = 0; l < LL; ++l) {
        const size_t odd = (size_t)l * DM * DM;
        const size_t odh = (size_t)l * DM * HIDD;
        convT_kernel<<<cdiv(DM * DM, 256), 256, 0, stream>>>(wq + odd, wqT + odd, DM, DM);
        convT_kernel<<<cdiv(DM * DM, 256), 256, 0, stream>>>(wk + odd, wkT + odd, DM, DM);
        convT_kernel<<<cdiv(DM * DM, 256), 256, 0, stream>>>(wv + odd, wvT + odd, DM, DM);
        convT_kernel<<<cdiv(DM * DM, 256), 256, 0, stream>>>(wo + odd, woT + odd, DM, DM);
        convT_kernel<<<cdiv(DM * HIDD, 256), 256, 0, stream>>>(w1 + odh, w1T + odh, DM, HIDD);
        convT_kernel<<<cdiv(HIDD * DM, 256), 256, 0, stream>>>(w2 + odh, w2T + odh, HIDD, DM);
    }
    convT_kernel<<<cdiv(DM * VV, 256), 256, 0, stream>>>(w_out, wOT, DM, VV);

    embed_kernel<<<cdiv(MM * DM, 256), 256, 0, stream>>>(tokens, emb, pe, xF);

    const float scale = 1.0f / sqrtf((float)DM);  // NOTE: sqrt(d_model) per reference

    // GEMM variants (fusion flags as template args; no runtime branches)
    auto gemm_proj = [&](const bf16* A, const bf16* BT, const float* bias,
                         bf16* oB, int Mm, int Nn, int Kk) {   // bias -> bf16 out
        dim3 g(Mm / 64, Nn / 128);
        gemm_bf16_kernel<true, false, false, false, true, false>
            <<<g, 256, 0, stream>>>(A, BT, bias, nullptr, nullptr, oB, Mm, Nn, Kk);
    };
    auto gemm_relu = [&](const bf16* A, const bf16* BT, const float* bias,
                         bf16* oB, int Mm, int Nn, int Kk) {   // bias+relu -> bf16
        dim3 g(Mm / 64, Nn / 128);
        gemm_bf16_kernel<true, false, true, false, true, false>
            <<<g, 256, 0, stream>>>(A, BT, bias, nullptr, nullptr, oB, Mm, Nn, Kk);
    };
    auto gemm_res = [&](const bf16* A, const bf16* BT, const float* bias,
                        const float* resid, float* oF, int Mm, int Nn, int Kk) {
        dim3 g(Mm / 64, Nn / 128);                             // bias+resid -> f32
        gemm_bf16_kernel<true, true, false, true, false, false>
            <<<g, 256, 0, stream>>>(A, BT, bias, resid, oF, nullptr, Mm, Nn, Kk);
    };
    auto gemm_logits = [&](const bf16* A, const bf16* BT, float* oF,
                           int Mm, int Nn, int Kk) {           // plain -> NT f32
        dim3 g(Mm / 64, Nn / 128);
        gemm_bf16_kernel<false, false, false, true, false, true>
            <<<g, 256, 0, stream>>>(A, BT, nullptr, nullptr, oF, nullptr, Mm, Nn, Kk);
    };

    for (int l = 0; l < LL; ++l) {
        const size_t od  = (size_t)l * DM;
        const size_t oh  = (size_t)l * HIDD;
        const size_t odd = (size_t)l * DM * DM;
        const size_t odh = (size_t)l * DM * HIDD;

        ln_kernel<<<MM, 256, 0, stream>>>(xF, ln1_g + od, ln1_b + od, hB);
        gemm_proj(hB, wqT + odd, bq + od, qB, MM, DM, DM);
        gemm_proj(hB, wkT + odd, bk + od, kB, MM, DM, DM);
        gemm_proj(hB, wvT + odd, bv + od, vB, MM, DM, DM);
        attn_kernel<<<dim3(SS / 16, HH, BB), 32, 0, stream>>>(qB, kB, vB, aB, scale);
        gemm_res(aB, woT + odd, bo + od, xF, xF, MM, DM, DM);

        ln_kernel<<<MM, 256, 0, stream>>>(xF, ln2_g + od, ln2_b + od, hB);
        gemm_relu(hB, w1T + odh, b1 + oh, fB, MM, HIDD, DM);
        gemm_res(fB, w2T + odh, b2 + od, xF, xF, MM, DM, HIDD);
    }

    ln_kernel<<<MM, 256, 0, stream>>>(xF, lnf_g, lnf_b, hB);
    gemm_logits(hB, wOT, logits, MM, VV, DM);
}
